// RoIHead_1606317769519
// MI455X (gfx1250) — compile-verified, matrix operand order checked by reading, not run on hardware
//
#include <hip/hip_runtime.h>
#include <math.h>

// ---------------------------------------------------------------------------
// Types for CDNA5 WMMA (wave32): D(16x16 f32) = A(16x32 f16) x B(32x16 f16) + C
// ---------------------------------------------------------------------------
typedef _Float16 h16;
typedef __attribute__((ext_vector_type(16))) _Float16     v16h;
typedef __attribute__((ext_vector_type(8)))  float        v8f;
typedef __attribute__((ext_vector_type(4)))  unsigned int u4v;

union Frag16 { v16h v; unsigned int u[8]; };

// Problem constants (match reference)
#define NRI   600      // N_ROIS
#define CIN   128
#define FH    38
#define FW    38
#define DFC   1024
#define NREL  16
#define DK    64
#define PK    (CIN*49) // 6272
#define NCLS  21
#define SCALE 0.0625f  // 1/16

// ---------------------------------------------------------------------------
// RoI max pool: out[roi][c*49 + ph*7 + pw]  (f16, ready as GEMM A operand)
// ---------------------------------------------------------------------------
__global__ void roi_pool_kernel(const float* __restrict__ x,
                                const float* __restrict__ rois,
                                const int* __restrict__ roi_idx,
                                h16* __restrict__ out)
{
    int roi = blockIdx.x;
    int c   = blockIdx.y;
    int t   = threadIdx.x;
    if (t >= 49) return;
    // ir permutation [0,2,1,4,3]: ir1=xmin=rois[:,1], ir2=ymin=rois[:,0],
    //                             ir3=xmax=rois[:,3], ir4=ymax=rois[:,2]
    float ymn = rois[roi * 4 + 0];
    float xmn = rois[roi * 4 + 1];
    float ymx = rois[roi * 4 + 2];
    float xmx = rois[roi * 4 + 3];
    int xs = (int)floorf(xmn * SCALE + 0.5f);
    int ys = (int)floorf(ymn * SCALE + 0.5f);
    int xe = (int)floorf(xmx * SCALE + 0.5f);
    int ye = (int)floorf(ymx * SCALE + 0.5f);
    float bw = fmaxf((float)(xe - xs + 1), 1.0f) * (1.0f / 7.0f);
    float bh = fmaxf((float)(ye - ys + 1), 1.0f) * (1.0f / 7.0f);
    int ph = t / 7, pw = t % 7;
    int ws = min(max((int)floorf((float)pw * bw) + xs, 0), FW);
    int we = min(max((int)ceilf((float)(pw + 1) * bw) + xs, 0), FW);
    int hs = min(max((int)floorf((float)ph * bh) + ys, 0), FH);
    int he = min(max((int)ceilf((float)(ph + 1) * bh) + ys, 0), FH);
    const float* xc = x + ((long)roi_idx[roi] * CIN + c) * (FH * FW);
    float mx = -1e30f;
    for (int h = hs; h < he; ++h)
        for (int w = ws; w < we; ++w)
            mx = fmaxf(mx, xc[h * FW + w]);
    if (we <= ws || he <= hs) mx = 0.0f;
    out[(long)roi * PK + c * 49 + t] = (h16)mx;
}

// ---------------------------------------------------------------------------
// Transpose + convert: in (R x C, f32) -> out (C x R, f16), optional batch
// ---------------------------------------------------------------------------
__global__ void transpose_h16_kernel(const float* __restrict__ in,
                                     h16* __restrict__ out,
                                     int R, int C, long sIn, long sOut)
{
    long b = blockIdx.z;
    const float* ip = in + b * sIn;
    h16* op = out + b * sOut;
    long i = (long)blockIdx.x * blockDim.x + threadIdx.x;
    if (i >= (long)R * C) return;
    int r = (int)(i / C), c = (int)(i % C);
    op[(long)c * R + r] = (h16)ip[i];
}

// ---------------------------------------------------------------------------
// Fused position embedding + wg logits:
//   out[r][m][n] = log(max(relu(sum_g emb(m,n,g)*WGw[r,g] + WGb[r]), 1e-6))
// ---------------------------------------------------------------------------
__global__ void poswg_kernel(const float* __restrict__ rois,
                             const float* __restrict__ wgw,
                             const float* __restrict__ wgb,
                             float* __restrict__ out)
{
    __shared__ float sW[NREL * DK];
    __shared__ float sB[NREL];
    for (int i = threadIdx.x; i < NREL * DK; i += blockDim.x) sW[i] = wgw[i];
    if (threadIdx.x < NREL) sB[threadIdx.x] = wgb[threadIdx.x];
    __syncthreads();

    long idx = (long)blockIdx.x * blockDim.x + threadIdx.x;
    if (idx >= (long)NRI * NRI) return;
    int m = (int)(idx / NRI), n = (int)(idx % NRI);

    float ym_m = rois[m*4+0], xm_m = rois[m*4+1], yx_m = rois[m*4+2], xx_m = rois[m*4+3];
    float ym_n = rois[n*4+0], xm_n = rois[n*4+1], yx_n = rois[n*4+2], xx_n = rois[n*4+3];
    float cxm = (xm_m + xx_m) * 0.5f, cym = (ym_m + yx_m) * 0.5f;
    float cxn = (xm_n + xx_n) * 0.5f, cyn = (ym_n + yx_n) * 0.5f;
    float wm = xx_m - xm_m + 1.0f, hm = yx_m - ym_m + 1.0f;
    float wn = xx_n - xm_n + 1.0f, hn = yx_n - ym_n + 1.0f;

    float pos4[4];
    pos4[0] = logf(fmaxf(fabsf((cxm - cxn) / wm), 0.001f));
    pos4[1] = logf(fmaxf(fabsf((cym - cyn) / hm), 0.001f));
    pos4[2] = logf(wm / wn);
    pos4[3] = logf(hm / hn);

    float emb[64];
    #pragma unroll
    for (int j = 0; j < 4; ++j) {
        #pragma unroll
        for (int t = 0; t < 8; ++t) {
            float dm  = powf(1000.0f, -(float)t * 0.125f);
            float arg = 100.0f * pos4[j] * dm;
            int g = j * 8 + t;
            emb[g]      = sinf(arg);
            emb[32 + g] = cosf(arg);
        }
    }
    for (int r = 0; r < NREL; ++r) {
        float a = sB[r];
        #pragma unroll 8
        for (int g = 0; g < 64; ++g) a += emb[g] * sW[r * 64 + g];
        a = fmaxf(a, 0.0f);
        out[((long)r * NRI + m) * NRI + n] = logf(fmaxf(a, 1e-6f));
    }
}

// ---------------------------------------------------------------------------
// Fragment loaders (16-bit A-layout; B stored N x K uses the same pattern).
// Fast path: two 16-byte vector loads, no predication (rows are pre-clamped,
// out-of-range row garbage only reaches stores that are masked anyway).
// ---------------------------------------------------------------------------
__device__ __forceinline__ void load_frag_fast(Frag16& f, const h16* __restrict__ rp,
                                               int kk, int half16)
{
    const u4v lo = *reinterpret_cast<const u4v*>(rp + kk + 8 * half16);
    const u4v hi = *reinterpret_cast<const u4v*>(rp + kk + 16 + 8 * half16);
    f.u[0] = lo[0]; f.u[1] = lo[1]; f.u[2] = lo[2]; f.u[3] = lo[3];
    f.u[4] = hi[0]; f.u[5] = hi[1]; f.u[6] = hi[2]; f.u[7] = hi[3];
}

// Tail path (K not multiple of 32): clamped-address loads + branchless select.
__device__ __forceinline__ void load_frag_tail(Frag16& f, const h16* __restrict__ rp,
                                               int kk, int K, int half16)
{
    #pragma unroll
    for (int p = 0; p < 8; ++p) {
        int k  = kk + ((p < 4) ? 0 : 16) + 8 * half16 + 2 * (p & 3);
        int kc = (k < K) ? k : 0;                 // K is even: k<K => k+1<K
        unsigned int d = *reinterpret_cast<const unsigned int*>(rp + kc);
        f.u[p] = (k < K) ? d : 0u;
    }
}

// ---------------------------------------------------------------------------
// Generic WMMA GEMM:  C(MxN,f32) = alpha*A(MxK,f16) * Bt(NxK,f16)^T
//                      [+ bias[n]] [+ resid[m][n]] [relu]
// optional f16 copy output, optional transposed store (writes [n][m]).
// Block = 128 threads (4 waves). Each wave computes a 16(M) x 64(N) strip
// (A fragment loaded once, reused for 4 WMMAs). Block covers 64M x 64N.
// Batched via blockIdx.z + strides.
// ---------------------------------------------------------------------------
__global__ void wmma_gemm_kernel(const h16* __restrict__ A, long sA, int lda,
                                 const h16* __restrict__ B, long sB, int ldb,
                                 float* __restrict__ C, long sC, int ldc,
                                 h16* __restrict__ Ch, long sCh, int ldch,
                                 const float* __restrict__ bias, long sBias,
                                 const float* __restrict__ resid, long sR, int ldr,
                                 int M, int N, int K,
                                 float alpha, int relu, int transOut)
{
    int lane   = threadIdx.x & 31;
    int wave   = threadIdx.x >> 5;          // 0..3 -> M sub-tile
    int b      = blockIdx.z;
    int mTile  = blockIdx.y * 64 + wave * 16;
    int nTile  = blockIdx.x * 64;
    int half16 = lane >> 4;
    int l15    = lane & 15;

    A += (long)b * sA;
    B += (long)b * sB;
    if (C)     C     += (long)b * sC;
    if (Ch)    Ch    += (long)b * sCh;
    if (bias)  bias  += (long)b * sBias;
    if (resid) resid += (long)b * sR;

    // clamped row pointers (always-safe unconditional loads)
    const h16* Ap = A + (long)min(mTile + l15, M - 1) * lda;
    const h16* Bp[4];
    #pragma unroll
    for (int j = 0; j < 4; ++j)
        Bp[j] = B + (long)min(nTile + j * 16 + l15, N - 1) * ldb;

    v8f acc[4] = {};
    int Kfull = K & ~31;
    for (int kk = 0; kk < Kfull; kk += 32) {
        Frag16 fa;
        load_frag_fast(fa, Ap, kk, half16);
        #pragma unroll
        for (int j = 0; j < 4; ++j) {
            Frag16 fb;
            load_frag_fast(fb, Bp[j], kk, half16);
            acc[j] = __builtin_amdgcn_wmma_f32_16x16x32_f16(
                false, fa.v, false, fb.v, (short)0, acc[j], false, false);
        }
    }
    if (Kfull < K) {
        Frag16 fa;
        load_frag_tail(fa, Ap, Kfull, K, half16);
        #pragma unroll
        for (int j = 0; j < 4; ++j) {
            Frag16 fb;
            load_frag_tail(fb, Bp[j], Kfull, K, half16);
            acc[j] = __builtin_amdgcn_wmma_f32_16x16x32_f16(
                false, fa.v, false, fb.v, (short)0, acc[j], false, false);
        }
    }

    #pragma unroll
    for (int j = 0; j < 4; ++j) {
        int n = nTile + j * 16 + l15;
        #pragma unroll
        for (int i = 0; i < 8; ++i) {
            int m = mTile + i + 8 * half16;
            if (m < M && n < N) {
                float v = acc[j][i] * alpha;
                if (bias)  v += bias[n];
                if (resid) v += resid[(long)m * ldr + n];
                if (relu)  v = fmaxf(v, 0.0f);
                if (!transOut) {
                    if (C)  C[(long)m * ldc + n]   = v;
                    if (Ch) Ch[(long)m * ldch + n] = (h16)v;
                } else {
                    if (C)  C[(long)n * ldc + m]   = v;
                    if (Ch) Ch[(long)n * ldch + m] = (h16)v;
                }
            }
        }
    }
}

// ---------------------------------------------------------------------------
// Softmax over n of (wg_log + scores); writes f16 attention.
// grid = (M, R), block = 256
// ---------------------------------------------------------------------------
__global__ void softmax_row_kernel(const float* __restrict__ scores,
                                   const float* __restrict__ wglog,
                                   h16* __restrict__ att, int Mrows, int N)
{
    int r = blockIdx.y, m = blockIdx.x;
    long off = ((long)r * Mrows + m) * N;
    const float* s = scores + off;
    const float* w = wglog + off;
    h16* o = att + off;

    __shared__ float red[256];
    float l[3];
    float mx = -1e30f;
    #pragma unroll
    for (int j = 0; j < 3; ++j) {
        int n = threadIdx.x + j * 256;
        l[j] = (n < N) ? (s[n] + w[n]) : -1e30f;
        mx = fmaxf(mx, l[j]);
    }
    red[threadIdx.x] = mx; __syncthreads();
    for (int st = 128; st > 0; st >>= 1) {
        if (threadIdx.x < st) red[threadIdx.x] = fmaxf(red[threadIdx.x], red[threadIdx.x + st]);
        __syncthreads();
    }
    mx = red[0]; __syncthreads();

    float e[3]; float sum = 0.0f;
    #pragma unroll
    for (int j = 0; j < 3; ++j) {
        int n = threadIdx.x + j * 256;
        e[j] = (n < N) ? expf(l[j] - mx) : 0.0f;
        sum += e[j];
    }
    red[threadIdx.x] = sum; __syncthreads();
    for (int st = 128; st > 0; st >>= 1) {
        if (threadIdx.x < st) red[threadIdx.x] += red[threadIdx.x + st];
        __syncthreads();
    }
    float inv = 1.0f / red[0];
    #pragma unroll
    for (int j = 0; j < 3; ++j) {
        int n = threadIdx.x + j * 256;
        if (n < N) o[n] = (h16)(e[j] * inv);
    }
}

// ---------------------------------------------------------------------------
// Host side
// ---------------------------------------------------------------------------
static void gemm(hipStream_t st,
                 const h16* A, long sA, int lda,
                 const h16* B, long sB, int ldb,
                 float* C, long sC, int ldc,
                 h16* Ch, long sCh, int ldch,
                 const float* bias, long sBias,
                 const float* resid, long sR, int ldr,
                 int M, int N, int K, float alpha, int relu, int transOut, int batch)
{
    dim3 g((N + 63) / 64, (M + 63) / 64, batch);
    wmma_gemm_kernel<<<g, dim3(128, 1, 1), 0, st>>>(
        A, sA, lda, B, sB, ldb, C, sC, ldc, Ch, sCh, ldch,
        bias, sBias, resid, sR, ldr, M, N, K, alpha, relu, transOut);
}

static void transpose_w(hipStream_t st, const float* in, h16* out,
                        int R, int C, long sIn, long sOut, int batch)
{
    long tot = (long)R * C;
    dim3 g((unsigned)((tot + 255) / 256), 1, batch);
    transpose_h16_kernel<<<g, 256, 0, st>>>(in, out, R, C, sIn, sOut);
}

extern "C" void kernel_launch(void* const* d_in, const int* in_sizes, int n_in,
                              void* d_out, int out_size, void* d_ws, size_t ws_size,
                              hipStream_t stream)
{
    (void)n_in; (void)out_size; (void)ws_size;
    const float* x      = (const float*)d_in[0];
    const float* rois   = (const float*)d_in[1];
    const int*   ridx   = (const int*)d_in[2];
    const float* fc1_w  = (const float*)d_in[3];
    const float* fc1_b  = (const float*)d_in[4];

    // Relation param dicts: detect flattening order at runtime.
    // alphabetical (jax tree sort): WG_b(16), WG_w, WK_b, WK_w, WQ_b, WQ_w, WV_b, WV_w
    // insertion order:              WG_w(1024), WG_b, WK_w, WK_b, WQ_w, WQ_b, WV_w, WV_b
    const float *wgw[2], *wgb[2], *wkw[2], *wkb[2], *wqw[2], *wqb[2], *wvw[2], *wvb[2];
    bool alpha_order = (in_sizes[5] == 16);
    for (int rset = 0; rset < 2; ++rset) {
        int o = 5 + rset * 8;
        if (alpha_order) {
            wgb[rset] = (const float*)d_in[o+0]; wgw[rset] = (const float*)d_in[o+1];
            wkb[rset] = (const float*)d_in[o+2]; wkw[rset] = (const float*)d_in[o+3];
            wqb[rset] = (const float*)d_in[o+4]; wqw[rset] = (const float*)d_in[o+5];
            wvb[rset] = (const float*)d_in[o+6]; wvw[rset] = (const float*)d_in[o+7];
        } else {
            wgw[rset] = (const float*)d_in[o+0]; wgb[rset] = (const float*)d_in[o+1];
            wkw[rset] = (const float*)d_in[o+2]; wkb[rset] = (const float*)d_in[o+3];
            wqw[rset] = (const float*)d_in[o+4]; wqb[rset] = (const float*)d_in[o+5];
            wvw[rset] = (const float*)d_in[o+6]; wvb[rset] = (const float*)d_in[o+7];
        }
    }
    const float* fc2_w   = (const float*)d_in[21];
    const float* fc2_b   = (const float*)d_in[22];
    const float* cls_w   = (const float*)d_in[23];
    const float* cls_b   = (const float*)d_in[24];
    const float* score_w = (const float*)d_in[25];
    const float* score_b = (const float*)d_in[26];
    float* out = (float*)d_out;

    // ---- workspace sub-allocation (256B aligned) ----
    char* base = (char*)d_ws;
    size_t off = 0;
    auto alloc = [&](size_t bytes) -> void* {
        void* p = base + off;
        off += (bytes + 255) & ~(size_t)255;
        return p;
    };
    h16*   poolH   = (h16*)alloc((size_t)NRI * PK * 2);
    h16*   fc1wT   = (h16*)alloc((size_t)PK * DFC * 2);
    h16*   fc2wT   = (h16*)alloc((size_t)DFC * DFC * 2);
    h16*   clswT   = (h16*)alloc((size_t)(NCLS * 4) * DFC * 2);
    h16*   scrwT   = (h16*)alloc((size_t)NCLS * DFC * 2);
    h16*   kqvT[2][3];
    for (int rset = 0; rset < 2; ++rset)
        for (int j = 0; j < 3; ++j)
            kqvT[rset][j] = (h16*)alloc((size_t)NREL * DK * DFC * 2);
    float* wglog   = (float*)alloc((size_t)NREL * NRI * NRI * 4);
    float* scores  = (float*)alloc((size_t)NREL * NRI * NRI * 4);
    h16*   att     = (h16*)alloc((size_t)NREL * NRI * NRI * 2);
    h16*   kH      = (h16*)alloc((size_t)NREL * NRI * DK * 2);
    h16*   qH      = (h16*)alloc((size_t)NREL * NRI * DK * 2);
    h16*   vTH     = (h16*)alloc((size_t)NREL * DK * NRI * 2);
    float* feat0   = (float*)alloc((size_t)NRI * DFC * 4);
    h16*   feat0H  = (h16*)alloc((size_t)NRI * DFC * 2);
    float* feat1   = (float*)alloc((size_t)NRI * DFC * 4);
    h16*   feat1H  = (h16*)alloc((size_t)NRI * DFC * 2);
    (void)alloc(256); // slack padding; tail loads are clamped to k < K

    // ---- weight conversion / transposition (f32 -> f16, stored N x K) ----
    transpose_w(stream, fc1_w, fc1wT, PK, DFC, 0, 0, 1);
    transpose_w(stream, fc2_w, fc2wT, DFC, DFC, 0, 0, 1);
    transpose_w(stream, cls_w, clswT, DFC, NCLS * 4, 0, 0, 1);
    transpose_w(stream, score_w, scrwT, DFC, NCLS, 0, 0, 1);
    for (int rset = 0; rset < 2; ++rset) {
        transpose_w(stream, wkw[rset], kqvT[rset][0], DFC, DK, (long)DFC*DK, (long)DK*DFC, NREL);
        transpose_w(stream, wqw[rset], kqvT[rset][1], DFC, DK, (long)DFC*DK, (long)DK*DFC, NREL);
        transpose_w(stream, wvw[rset], kqvT[rset][2], DFC, DK, (long)DFC*DK, (long)DK*DFC, NREL);
    }

    // ---- RoI pooling ----
    roi_pool_kernel<<<dim3(NRI, CIN), 64, 0, stream>>>(x, rois, ridx, poolH);

    // ---- fc1 = relu(pool @ fc1_w + b) ----
    gemm(stream, poolH, 0, PK, fc1wT, 0, PK,
         feat0, 0, DFC, feat0H, 0, DFC,
         fc1_b, 0, nullptr, 0, 0,
         NRI, DFC, PK, 1.0f, 1, 0, 1);

    const float rsqrtK = 0.125f; // 1/sqrt(64)
    long posBlocks = ((long)NRI * NRI + 255) / 256;

    for (int rset = 0; rset < 2; ++rset) {
        float* fin  = feat0;  h16* finH  = feat0H;   // relation input (f32 + f16)
        float* fout = feat1;  h16* foutH = feat1H;   // relation output

        // wg logits (pos embedding fused)
        poswg_kernel<<<(unsigned)posBlocks, 256, 0, stream>>>(rois, wgw[rset], wgb[rset], wglog);

        // K, Q, V (batched over 16 relations)
        gemm(stream, finH, 0, DFC, kqvT[rset][0], (long)DK*DFC, DFC,
             nullptr, 0, 0, kH, (long)NRI*DK, DK,
             wkb[rset], DK, nullptr, 0, 0,
             NRI, DK, DFC, 1.0f, 0, 0, NREL);
        gemm(stream, finH, 0, DFC, kqvT[rset][1], (long)DK*DFC, DFC,
             nullptr, 0, 0, qH, (long)NRI*DK, DK,
             wqb[rset], DK, nullptr, 0, 0,
             NRI, DK, DFC, 1.0f, 0, 0, NREL);
        gemm(stream, finH, 0, DFC, kqvT[rset][2], (long)DK*DFC, DFC,  // V, stored transposed
             nullptr, 0, 0, vTH, (long)DK*NRI, NRI,
             wvb[rset], DK, nullptr, 0, 0,
             NRI, DK, DFC, 1.0f, 0, 1, NREL);

        // scores[r] = k @ q^T / sqrt(64)
        gemm(stream, kH, (long)NRI*DK, DK, qH, (long)NRI*DK, DK,
             scores, (long)NRI*NRI, NRI, nullptr, 0, 0,
             nullptr, 0, nullptr, 0, 0,
             NRI, NRI, DK, rsqrtK, 0, 0, NREL);

        // att = softmax(wg_log + scores)
        softmax_row_kernel<<<dim3(NRI, NREL), 256, 0, stream>>>(scores, wglog, att, NRI, NRI);

        // out[:, r*64:(r+1)*64] = att[r] @ v[r] + residual(fin)
        gemm(stream, att, (long)NRI*NRI, NRI, vTH, (long)DK*NRI, NRI,
             fout, DK, DFC, foutH, DK, DFC,
             nullptr, 0, fin, DK, DFC,
             NRI, DK, NRI, 1.0f, 0, 0, NREL);

        if (rset == 0) {
            // fc2 = relu(rel1_out @ fc2_w + b)  -> becomes input of relation 2
            gemm(stream, feat1H, 0, DFC, fc2wT, 0, DFC,
                 feat0, 0, DFC, feat0H, 0, DFC,
                 fc2_b, 0, nullptr, 0, 0,
                 NRI, DFC, DFC, 1.0f, 1, 0, 1);
        }
    }

    // ---- heads: cls (600x84) then score (600x21), concatenated in d_out ----
    gemm(stream, feat1H, 0, DFC, clswT, 0, DFC,
         out, 0, NCLS * 4, nullptr, 0, 0,
         cls_b, 0, nullptr, 0, 0,
         NRI, NCLS * 4, DFC, 1.0f, 0, 0, 1);
    gemm(stream, feat1H, 0, DFC, scrwT, 0, DFC,
         out + (size_t)NRI * NCLS * 4, 0, NCLS, nullptr, 0, 0,
         score_b, 0, nullptr, 0, 0,
         NRI, NCLS, DFC, 1.0f, 0, 0, 1);
}
